// SelfDotProductAttention_9543417332476
// MI455X (gfx1250) — compile-verified, matrix operand order checked by reading, not run on hardware
//
#include <hip/hip_runtime.h>

// ---------------------------------------------------------------------------
// CDNA5 (gfx1250) self-attention pipeline, all GEMMs on v_wmma_f32_16x16x32_bf16
//   B=4, S=2048, D=1024  (single head, scores = Kproj . Qproj^T / 32)
// Double-buffered LDS staging via GLOBAL_LOAD_ASYNC_TO_LDS_B128 (ASYNCcnt),
// sync fallback if the builtin is absent. Builtin takes (AS1 v4i*, AS3 v4i*, imm, imm).
// ---------------------------------------------------------------------------

typedef __attribute__((ext_vector_type(16))) __bf16 v16bf;
typedef __attribute__((ext_vector_type(8)))  float  v8f;
typedef __attribute__((ext_vector_type(4)))  int    v4i;

#if defined(__AMDGCN__) && __has_builtin(__builtin_amdgcn_global_load_async_to_lds_b128)
#define ASYNC_LDS 1
#else
#define ASYNC_LDS 0
#endif

#if ASYNC_LDS
typedef __attribute__((address_space(1))) void gas_void;
typedef __attribute__((address_space(3))) void las_void;
typedef __attribute__((address_space(1))) v4i  gas_v4i;
typedef __attribute__((address_space(3))) v4i  las_v4i;
#define GLD_ASYNC_B128(dst_lds, src_glb)                                     \
    __builtin_amdgcn_global_load_async_to_lds_b128(                          \
        (gas_v4i*)(gas_void*)(src_glb), (las_v4i*)(las_void*)(dst_lds), 0, 0)
#endif

__device__ __forceinline__ void wait_async0() {
#if defined(__AMDGCN__)
#if __has_builtin(__builtin_amdgcn_s_wait_asynccnt)
    __builtin_amdgcn_s_wait_asynccnt(0);
#else
    asm volatile("s_wait_asynccnt 0x0" ::: "memory");
#endif
#endif
}

union Frag {
    v16bf v;
    uint4 q[2];
};

__device__ __forceinline__ unsigned short f2bf(float f) {
    unsigned u = __float_as_uint(f);
    unsigned r = ((u >> 16) & 1u) + 0x7FFFu;   // round-to-nearest-even
    return (unsigned short)((u + r) >> 16);
}

// ------------------------------ fp32 -> bf16 -------------------------------
__global__ __launch_bounds__(256)
void cvt_f32_bf16_x4(const float* __restrict__ in, unsigned short* __restrict__ out, int n4) {
    int i = blockIdx.x * blockDim.x + threadIdx.x;
    if (i < n4) {
        float4 f = ((const float4*)in)[i];
        ushort4 o;
        o.x = f2bf(f.x); o.y = f2bf(f.y); o.z = f2bf(f.z); o.w = f2bf(f.w);
        ((ushort4*)out)[i] = o;
    }
}

// ------------------------------ WMMA GEMM ----------------------------------
// C[m,n] = scale * sum_k A[m,k] * Bt[n,k]   (+ bias / transposed store variants)
// Block: 256 threads = 8 waves. Tile 128(M) x 128(N), K-step 32, double buffer.
// Wave w: rows (w&3)*32..+32, cols (w>>2)*64..+64  -> 2x4 subtiles of 16x16.
// OMODE: 0 = bf16 row-major   1 = bf16 transposed (C[n*ldc+m], vectorized)
//        2 = f32 row-major * scale        3 = f32 row-major + bias[n]
template <int OMODE>
__global__ __launch_bounds__(256)
void gemm_bt_wmma(const unsigned short* __restrict__ A,
                  const unsigned short* __restrict__ Bt,
                  void* __restrict__ Cv,
                  const float* __restrict__ bias,
                  int K, int lda, int ldb, int ldc, float scale,
                  size_t aBS, size_t bBS, size_t cBS) {
    __shared__ unsigned short As[2][128][40];  // 32 + 8 pad: 80B pitch -> conflict-free b128 frags
    __shared__ unsigned short Bs[2][128][40];

    const int t    = threadIdx.x;
    const int lane = t & 31;
    const int w    = t >> 5;
    const int z    = blockIdx.z;
    const int m0   = blockIdx.y * 128;
    const int n0   = blockIdx.x * 128;

    // global->LDS staging: thread -> (row 0..127, 16-col half)
    const int lrow = t >> 1;
    const int lcol = (t & 1) * 16;
    const unsigned short* aSrc = A + aBS * z + (size_t)(m0 + lrow) * lda + lcol;
    const unsigned short* bSrc = Bt + bBS * z + (size_t)(n0 + lrow) * ldb + lcol;

    const int laneM = lane & 15;
    const int hi8   = (lane >> 4) * 8;
    const int wm    = (w & 3) * 32;
    const int wn    = (w >> 2) * 64;

    v8f acc[2][4];
    #pragma unroll
    for (int i = 0; i < 2; ++i)
        #pragma unroll
        for (int j = 0; j < 4; ++j)
            #pragma unroll
            for (int e = 0; e < 8; ++e) acc[i][j][e] = 0.0f;

    auto stage = [&](int kk, int pb) {
#if ASYNC_LDS
        GLD_ASYNC_B128(&As[pb][lrow][lcol],     aSrc + kk);
        GLD_ASYNC_B128(&As[pb][lrow][lcol + 8], aSrc + kk + 8);
        GLD_ASYNC_B128(&Bs[pb][lrow][lcol],     bSrc + kk);
        GLD_ASYNC_B128(&Bs[pb][lrow][lcol + 8], bSrc + kk + 8);
#else
        *(uint4*)&As[pb][lrow][lcol]     = *(const uint4*)(aSrc + kk);
        *(uint4*)&As[pb][lrow][lcol + 8] = *(const uint4*)(aSrc + kk + 8);
        *(uint4*)&Bs[pb][lrow][lcol]     = *(const uint4*)(bSrc + kk);
        *(uint4*)&Bs[pb][lrow][lcol + 8] = *(const uint4*)(bSrc + kk + 8);
#endif
    };

    const int nsteps = K >> 5;
    stage(0, 0);                                  // prologue: tile 0 in flight

    for (int i = 0; i < nsteps; ++i) {
        const int pb = i & 1;
#if ASYNC_LDS
        wait_async0();                            // my async copies for buf[pb] done
#endif
        __syncthreads();                          // everyone's copies visible; prev buf free

        if (i + 1 < nsteps) {                     // overlap next tile's copy with compute
            stage((i + 1) * 32, pb ^ 1);
            if (i + 2 < nsteps) {                 // warm L2 one slab further ahead
                __builtin_prefetch(aSrc + (i + 2) * 32, 0, 1);
                __builtin_prefetch(bSrc + (i + 2) * 32, 0, 1);
            }
        }

        Frag fa[2], fb[4];
        #pragma unroll
        for (int sm = 0; sm < 2; ++sm) {
            const unsigned short* p = &As[pb][wm + sm * 16 + laneM][0];
            fa[sm].q[0] = *(const uint4*)(p + hi8);
            fa[sm].q[1] = *(const uint4*)(p + 16 + hi8);
        }
        #pragma unroll
        for (int sn = 0; sn < 4; ++sn) {
            const unsigned short* p = &Bs[pb][wn + sn * 16 + laneM][0];
            fb[sn].q[0] = *(const uint4*)(p + hi8);
            fb[sn].q[1] = *(const uint4*)(p + 16 + hi8);
        }
        #pragma unroll
        for (int sm = 0; sm < 2; ++sm)
            #pragma unroll
            for (int sn = 0; sn < 4; ++sn)
                acc[sm][sn] = __builtin_amdgcn_wmma_f32_16x16x32_bf16(
                    false, fa[sm].v, false, fb[sn].v,
                    (short)0, acc[sm][sn], false, false);
    }

    // ------------------------------ epilogue -------------------------------
    #pragma unroll
    for (int sm = 0; sm < 2; ++sm) {
        #pragma unroll
        for (int sn = 0; sn < 4; ++sn) {
            const int n    = n0 + wn + sn * 16 + laneM;
            const int mrow = m0 + wm + sm * 16 + hi8;   // 8 consecutive M per lane
            if (OMODE == 0) {
                unsigned short* C = (unsigned short*)Cv + cBS * z;
                #pragma unroll
                for (int r = 0; r < 8; ++r)
                    C[(size_t)(mrow + r) * ldc + n] = f2bf(acc[sm][sn][r]);
            } else if (OMODE == 1) {
                unsigned short* C = (unsigned short*)Cv;
                alignas(16) unsigned short tmp[8];
                #pragma unroll
                for (int r = 0; r < 8; ++r) tmp[r] = f2bf(acc[sm][sn][r]);
                *(uint4*)&C[(size_t)n * ldc + mrow] = *(const uint4*)tmp;
            } else if (OMODE == 2) {
                float* C = (float*)Cv + cBS * z;
                #pragma unroll
                for (int r = 0; r < 8; ++r)
                    C[(size_t)(mrow + r) * ldc + n] = acc[sm][sn][r] * scale;
            } else {
                float* C = (float*)Cv;
                const float bv = bias[n];
                #pragma unroll
                for (int r = 0; r < 8; ++r)
                    C[(size_t)(mrow + r) * ldc + n] = acc[sm][sn][r] + bv;
            }
        }
    }
}

// ------------------------------ row softmax --------------------------------
// one block per row of 2048 fp32 logits -> bf16 probabilities
__global__ __launch_bounds__(256)
void softmax_row2048(const float* __restrict__ Sc, unsigned short* __restrict__ P) {
    __shared__ float red[256];
    const size_t row = blockIdx.x;
    const float* src = Sc + row * 2048;
    unsigned short* dst = P + row * 2048;
    const int t = threadIdx.x;

    float v[8];
    float m = -3.402823466e38f;
    #pragma unroll
    for (int i = 0; i < 8; ++i) { v[i] = src[t + i * 256]; m = fmaxf(m, v[i]); }
    red[t] = m; __syncthreads();
    for (int s = 128; s > 0; s >>= 1) {
        if (t < s) red[t] = fmaxf(red[t], red[t + s]);
        __syncthreads();
    }
    m = red[0]; __syncthreads();

    float e[8], sum = 0.f;
    #pragma unroll
    for (int i = 0; i < 8; ++i) { e[i] = __expf(v[i] - m); sum += e[i]; }
    red[t] = sum; __syncthreads();
    for (int s = 128; s > 0; s >>= 1) {
        if (t < s) red[t] += red[t + s];
        __syncthreads();
    }
    const float inv = 1.0f / red[0];
    #pragma unroll
    for (int i = 0; i < 8; ++i) dst[t + i * 256] = f2bf(e[i] * inv);
}

// ---------------------------------------------------------------------------
extern "C" void kernel_launch(void* const* d_in, const int* in_sizes, int n_in,
                              void* d_out, int out_size, void* d_ws, size_t ws_size,
                              hipStream_t stream) {
    (void)in_sizes; (void)n_in; (void)out_size; (void)ws_size;
    constexpr int Bn = 4, S = 2048, D = 1024;
    constexpr size_t BS = (size_t)Bn * S;               // 8192 rows

    const float* x  = (const float*)d_in[0];
    const float* Wq = (const float*)d_in[1];
    const float* Wk = (const float*)d_in[2];
    const float* Wv = (const float*)d_in[3];
    const float* Wo = (const float*)d_in[4];
    const float* bo = (const float*)d_in[5];

    // ---- workspace layout (~168 MiB) ----
    char* p = (char*)d_ws;
    unsigned short* xh  = (unsigned short*)p; p += BS * D * 2;          // x   bf16 [8192,1024]
    unsigned short* qh  = (unsigned short*)p; p += BS * D * 2;          // Q   bf16
    unsigned short* kh  = (unsigned short*)p; p += BS * D * 2;          // K   bf16
    unsigned short* vT  = (unsigned short*)p; p += (size_t)D * BS * 2;  // V^T bf16 [1024,8192]
    unsigned short* wqh = (unsigned short*)p; p += (size_t)D * D * 2;
    unsigned short* wkh = (unsigned short*)p; p += (size_t)D * D * 2;
    unsigned short* wvh = (unsigned short*)p; p += (size_t)D * D * 2;
    unsigned short* woh = (unsigned short*)p; p += (size_t)D * D * 2;
    float*          sc  = (float*)p;          p += (size_t)Bn * S * S * 4;  // scores fp32
    unsigned short* at  = (unsigned short*)p; p += (size_t)Bn * S * S * 2;  // probs bf16
    unsigned short* sv  = xh;   // alias: x_bf16 dead once Q/K/V exist

    const dim3 blk(256);

    // 1) convert inputs to bf16
    cvt_f32_bf16_x4<<<dim3((unsigned)(BS * D / 4 / 256)), blk, 0, stream>>>(x, xh, (int)(BS * D / 4));
    cvt_f32_bf16_x4<<<dim3(D * D / 4 / 256), blk, 0, stream>>>(Wq, wqh, D * D / 4);
    cvt_f32_bf16_x4<<<dim3(D * D / 4 / 256), blk, 0, stream>>>(Wk, wkh, D * D / 4);
    cvt_f32_bf16_x4<<<dim3(D * D / 4 / 256), blk, 0, stream>>>(Wv, wvh, D * D / 4);
    cvt_f32_bf16_x4<<<dim3(D * D / 4 / 256), blk, 0, stream>>>(Wo, woh, D * D / 4);

    // 2) projections: Q = x Wq^T, K = x Wk^T (bf16), V stored transposed [D, B*S]
    const dim3 gProj(D / 128, (unsigned)(BS / 128), 1);
    gemm_bt_wmma<0><<<gProj, blk, 0, stream>>>(xh, wqh, qh, nullptr, D, D, D, D, 1.f, 0, 0, 0);
    gemm_bt_wmma<0><<<gProj, blk, 0, stream>>>(xh, wkh, kh, nullptr, D, D, D, D, 1.f, 0, 0, 0);
    gemm_bt_wmma<1><<<gProj, blk, 0, stream>>>(xh, wvh, vT, nullptr, D, D, D, (int)BS, 1.f, 0, 0, 0);

    // 3) scores[b,i,j] = (K_i . Q_j) / 32   (fp32)
    const dim3 gSc(S / 128, S / 128, Bn);
    gemm_bt_wmma<2><<<gSc, blk, 0, stream>>>(kh, qh, sc, nullptr, D, D, D, S, 0.03125f,
                                             (size_t)S * D, (size_t)S * D, (size_t)S * S);

    // 4) row softmax -> bf16 probabilities
    softmax_row2048<<<dim3((unsigned)BS), blk, 0, stream>>>(sc, at);

    // 5) sum_v = attn @ V   (A = probs bf16, Bt row v = vT[v, b*S + j])
    const dim3 gAV(D / 128, S / 128, Bn);
    gemm_bt_wmma<0><<<gAV, blk, 0, stream>>>(at, vT, sv, nullptr, S, S, (int)BS, D, 1.f,
                                             (size_t)S * S, (size_t)S, (size_t)S * D);

    // 6) out = sum_v @ Wo^T + bo  (fp32 to d_out)
    gemm_bt_wmma<3><<<gProj, blk, 0, stream>>>(sv, woh, d_out, bo, D, D, D, D, 1.f, 0, 0, 0);
}
